// GaussianImage_RS_45896020525490
// MI455X (gfx1250) — matrix operand order repaired and checked by reading, not compiled, not run on hardware
//
#include <hip/hip_runtime.h>
#include <hip/hip_bf16.h>
#include <math.h>

typedef __attribute__((ext_vector_type(16))) _Float16 v16h;
typedef __attribute__((ext_vector_type(8)))  float    v8f;

#define IMG_H 512
#define IMG_W 512
#define TILE_W 16
#define TILE_H 4
#define TX (IMG_W / TILE_W)     // 32 tiles across
#define TY (IMG_H / TILE_H)     // 128 tiles down
#define NBINS (TX * TY)         // 4096
#define MAXE  (1 << 19)         // list capacity (expected ~30-60K entries)
#define ATH   (1.0f / 255.0f)
#define TWOPI 6.283185307179586f
#define LOG2E 1.4426950408889634f

// ---------------------------------------------------------------------------
// Kernel 0: zero bin counts.
// ---------------------------------------------------------------------------
__global__ void gs_zero_kernel(unsigned* __restrict__ p, int n) {
    int i = blockIdx.x * blockDim.x + threadIdx.x;
    if (i < n) p[i] = 0u;
}

// ---------------------------------------------------------------------------
// Kernel 1: per-gaussian preprocessing + exact-support tile rect + bin counts.
//   params[g] (12 used floats, stride 16): { cx, cy, hA, hB, hC, thr, op,
//                                            colR, colG, colB, 0, 0 }
//     hA = 0.5*A*log2e, hB = B*log2e, hC = 0.5*C*log2e  (exp -> exp2 domain)
//     thr = ATH/op so "alpha > ATH" <=> "e2 > thr"
//   Support radius (exact cull): alpha > ATH needs sigma < ln(255*op) and
//   sigma >= 0.5*|d|^2/lambda_max  =>  r = sqrt(2*lmax*ln(255*op)) (+1 pad).
//   params[nG] is a dummy entry (thr = +inf, colors 0) for chunk padding.
// ---------------------------------------------------------------------------
__global__ void gs_prep_kernel(const float* __restrict__ xyz,
                               const float* __restrict__ scaling,
                               const float* __restrict__ rotation,
                               const float* __restrict__ fdc,
                               const float* __restrict__ opacity,
                               float* __restrict__ params,
                               int4* __restrict__ rect,
                               unsigned* __restrict__ counts,
                               int nG) {
    int g = blockIdx.x * blockDim.x + threadIdx.x;
    if (g > nG) return;

    float* p = params + (size_t)g * 16;
    if (g == nG) {                       // dummy padding entry
        #pragma unroll
        for (int i = 0; i < 12; ++i) p[i] = 0.0f;
        p[5] = 3.4e38f;                  // thr = +inf -> alpha always culled
        return;
    }

    float mx = tanhf(xyz[2 * g + 0]);
    float my = tanhf(xyz[2 * g + 1]);
    float sx = fabsf(scaling[2 * g + 0] + 0.5f);
    float sy = fabsf(scaling[2 * g + 1] + 0.5f);
    float theta = (1.0f / (1.0f + __expf(-rotation[g]))) * TWOPI;
    float cx = 0.5f * ((mx + 1.0f) * (float)IMG_W - 1.0f);
    float cy = 0.5f * ((my + 1.0f) * (float)IMG_H - 1.0f);
    float cth = __cosf(theta), sth = __sinf(theta);
    float sx2 = sx * sx, sy2 = sy * sy;
    float a = cth * cth * sx2 + sth * sth * sy2;
    float b = cth * sth * (sx2 - sy2);
    float c = sth * sth * sx2 + cth * cth * sy2;
    float det = a * c - b * b;
    bool  valid = det > 0.0f;
    float inv_det = valid ? (1.0f / fmaxf(det, 1e-12f)) : 0.0f;
    float A  =  c * inv_det;
    float Bv = -b * inv_det;
    float C  =  a * inv_det;
    float op_eff = valid ? opacity[g] : 0.0f;
    float thr = (op_eff > 0.0f) ? (ATH / op_eff) : 3.4e38f;

    p[0] = cx;  p[1] = cy;
    p[2] = 0.5f * A * LOG2E;  p[3] = Bv * LOG2E;  p[4] = 0.5f * C * LOG2E;
    p[5] = thr; p[6] = op_eff;
    p[7] = fdc[3 * g + 0]; p[8] = fdc[3 * g + 1]; p[9] = fdc[3 * g + 2];
    p[10] = 0.0f; p[11] = 0.0f;

    // Exact support: lambda_max of covariance [[a,b],[b,c]].
    float lmax = 0.5f * (a + c) + sqrtf(fmaxf(0.25f * (a - c) * (a - c)
                                              + b * b, 0.0f));
    float q = op_eff * 255.0f;
    bool on = valid && (q > 1.0f);
    int4 rc; rc.x = 1; rc.y = 0; rc.z = 1; rc.w = 0;   // empty
    if (on) {
        float r = sqrtf(fmaxf(2.0f * lmax * __logf(q), 0.0f)) + 1.0f;
        float x0f = cx - r, x1f = cx + r, y0f = cy - r, y1f = cy + r;
        if (x1f >= 0.0f && x0f <= (float)(IMG_W - 1) &&
            y1f >= 0.0f && y0f <= (float)(IMG_H - 1)) {
            rc.x = max(0, (int)floorf(x0f * (1.0f / TILE_W)));
            rc.y = min(TX - 1, (int)floorf(x1f * (1.0f / TILE_W)));
            rc.z = max(0, (int)floorf(y0f * (1.0f / TILE_H)));
            rc.w = min(TY - 1, (int)floorf(y1f * (1.0f / TILE_H)));
            for (int yy = rc.z; yy <= rc.w; ++yy)
                for (int xx = rc.x; xx <= rc.y; ++xx)
                    atomicAdd(&counts[yy * TX + xx], 1u);
        }
    }
    rect[g] = rc;
}

// ---------------------------------------------------------------------------
// Kernel 2: exclusive prefix sum over 4096 bin counts (one block, LDS scan).
// Writes starts[] and an identical cursors[] for the scatter pass.
// ---------------------------------------------------------------------------
__global__ __launch_bounds__(1024) void gs_scan_kernel(
        const unsigned* __restrict__ counts,
        unsigned* __restrict__ starts,
        unsigned* __restrict__ cursors) {
    __shared__ unsigned sh[1024];
    const int tid = threadIdx.x;
    unsigned c[4], sum = 0;
    #pragma unroll
    for (int k = 0; k < 4; ++k) { c[k] = counts[tid * 4 + k]; sum += c[k]; }
    sh[tid] = sum;
    __syncthreads();
    for (int off = 1; off < 1024; off <<= 1) {
        unsigned v = (tid >= off) ? sh[tid - off] : 0u;
        __syncthreads();
        sh[tid] += v;
        __syncthreads();
    }
    unsigned excl = sh[tid] - sum;
    #pragma unroll
    for (int k = 0; k < 4; ++k) {
        starts[tid * 4 + k]  = excl;
        cursors[tid * 4 + k] = excl;
        excl += c[k];
    }
}

// ---------------------------------------------------------------------------
// Kernel 3: scatter gaussian indices into per-bin lists.
// ---------------------------------------------------------------------------
__global__ void gs_scatter_kernel(const int4* __restrict__ rect,
                                  unsigned* __restrict__ cursors,
                                  unsigned* __restrict__ list,
                                  int nG) {
    int g = blockIdx.x * blockDim.x + threadIdx.x;
    if (g >= nG) return;
    int4 rc = rect[g];
    for (int yy = rc.z; yy <= rc.w; ++yy)
        for (int xx = rc.x; xx <= rc.y; ++xx) {
            unsigned pos = atomicAdd(&cursors[yy * TX + xx], 1u);
            if (pos < MAXE) list[pos] = (unsigned)g;
        }
}

// ---------------------------------------------------------------------------
// Kernel 4: render. One wave per 16x4 pixel tile; 8 waves (8 bins) per block.
// Waves are fully independent (divergent chunk counts -> no block barriers;
// wave-local LDS RAW ordered with s_wait_dscnt). Per 32-entry chunk: gather
// params into wave-private LDS, build B tile (colors) and 4 A tiles (alphas,
// exp2 domain, fused f16 pack), accumulate via v_wmma_f32_16x16x32_f16.
// ---------------------------------------------------------------------------
__global__ __launch_bounds__(256) void gs_render_kernel(
        const float* __restrict__ params,
        const unsigned* __restrict__ counts,
        const unsigned* __restrict__ starts,
        const unsigned* __restrict__ list,
        float* __restrict__ out,
        int nG) {
    __shared__ float4 ldsE[8][96];        // per wave: 32 entries x 12 floats

    const int tid   = threadIdx.x;
    const int wave  = tid >> 5;
    const int lane  = tid & 31;
    const int grp   = lane >> 4;          // A/B/D lane-group
    const int mrow  = lane & 15;          // x offset within tile (A row M)
    const int nchan = lane & 15;          // D column (channel)
    const int bin   = blockIdx.x * 8 + wave;
    const int txi   = bin % TX;
    const int tyi   = bin / TX;

    const float pxf = (float)(txi * TILE_W + mrow);
    float pyf[4];
    #pragma unroll
    for (int t = 0; t < 4; ++t) pyf[t] = (float)(tyi * TILE_H + t);

    const unsigned cnt   = counts[bin];
    const unsigned start = starts[bin];
    const int nch = (int)((cnt + 31u) >> 5);

    v8f acc[4];
    #pragma unroll
    for (int t = 0; t < 4; ++t) { v8f z = {}; acc[t] = z; }

    float4* myE = ldsE[wave];
    const float4* myE4 = (const float4*)myE;
    const float*  lf   = (const float*)myE;

    for (int ch = 0; ch < nch; ++ch) {
        // ---- stage 32 entries (one per lane) into wave-private LDS ----
        int j = ch * 32 + lane;
        unsigned pos = start + (unsigned)j;
        int idx = nG;                                   // dummy by default
        if (j < (int)cnt && pos < MAXE) {
            unsigned v = list[pos];
            idx = (v < (unsigned)nG) ? (int)v : nG;
        }
        const float4* src = (const float4*)(params + (size_t)idx * 16);
        float4 a0 = src[0], a1 = src[1], a2 = src[2];
        myE[lane * 3 + 0] = a0;
        myE[lane * 3 + 1] = a1;
        myE[lane * 3 + 2] = a2;
        asm volatile("s_wait_dscnt 0x0" ::: "memory");  // wave-local RAW

        // ---- B tile: colors, contiguous K order (lane = grp*16 + N, elem = K&15)
        v16h bt;
        const int ebase = grp * 16;
        const int cidx = (nchan < 3) ? nchan : 0;
        #pragma unroll
        for (int e = 0; e < 16; ++e) {
            float cval = lf[(ebase + e) * 12 + 7 + cidx];
            bt[e] = (_Float16)((nchan < 3) ? cval : 0.0f);
        }

        // ---- A tiles: alphas. x fixed per lane -> dx terms hoisted per K.
        v16h at[4];
        #pragma unroll
        for (int s = 0; s < 16; ++s) {
            const int kk = grp * 16 + s;
            const float4 q0 = myE4[kk * 3 + 0];   // cx, cy, hA, hB
            const float4 q1 = myE4[kk * 3 + 1];   // hC, thr, op, colR
            const float dx = q0.x - pxf;
            const float u  = q0.z * dx * dx;      // hA*dx^2
            const float w  = q0.w * dx;           // hB*dx
            #pragma unroll
            for (int t = 0; t < 4; ++t) {
                float dy  = q0.y - pyf[t];
                float sig = fmaf(fmaf(q1.x, dy, w), dy, u);
                float e2  = __builtin_amdgcn_exp2f(-sig);
                _Float16 hv = (_Float16)(q1.z * e2);    // op*e (fma_mixhi)
                bool keep = (sig >= 0.0f) && (e2 > q1.y);
                at[t][s] = keep ? hv : (_Float16)0.0f;
            }
        }

        #pragma unroll
        for (int t = 0; t < 4; ++t) {
            acc[t] = __builtin_amdgcn_wmma_f32_16x16x32_f16(
                false, at[t], false, bt, (short)0, acc[t], false, false);
        }
    }

    // Writeout: D layout -> lanes hold N (channel), VGPR r holds M=r (lanes
    // 0-15) or M=8+r (lanes 16-31). Output is (1, 3, H, W). Empty bins write
    // zeros (output is poisoned by the harness).
    if (nchan < 3) {
        #pragma unroll
        for (int t = 0; t < 4; ++t) {
            int y = tyi * TILE_H + t;
            #pragma unroll
            for (int r = 0; r < 8; ++r) {
                int M = grp * 8 + r;
                int x = txi * TILE_W + M;
                float v = acc[t][r];
                v = fminf(fmaxf(v, 0.0f), 1.0f);
                out[(size_t)nchan * (IMG_H * IMG_W) + (size_t)y * IMG_W + x] = v;
            }
        }
    }
}

// ---------------------------------------------------------------------------
extern "C" void kernel_launch(void* const* d_in, const int* in_sizes, int n_in,
                              void* d_out, int out_size, void* d_ws,
                              size_t ws_size, hipStream_t stream) {
    const float* xyz      = (const float*)d_in[0];
    const float* scaling  = (const float*)d_in[1];
    const float* rotation = (const float*)d_in[2];
    const float* fdc      = (const float*)d_in[3];
    const float* opacity  = (const float*)d_in[4];

    const int N = in_sizes[0] / 2;              // (N,2) positions

    // Workspace layout (256B-aligned sections).
    char* ws = (char*)d_ws;
    size_t off = 0;
    float* params = (float*)(ws + off);         // (N+1) * 16 floats
    off += ((size_t)(N + 1) * 64 + 255) & ~(size_t)255;
    int4* rect = (int4*)(ws + off);             // N * int4
    off += ((size_t)N * 16 + 255) & ~(size_t)255;
    unsigned* counts = (unsigned*)(ws + off);   off += (size_t)NBINS * 4;
    unsigned* starts = (unsigned*)(ws + off);   off += (size_t)NBINS * 4;
    unsigned* cursors = (unsigned*)(ws + off);  off += (size_t)NBINS * 4;
    unsigned* list = (unsigned*)(ws + off);     // MAXE entries

    gs_zero_kernel<<<(NBINS + 255) / 256, 256, 0, stream>>>(counts, NBINS);
    gs_prep_kernel<<<(N + 1 + 255) / 256, 256, 0, stream>>>(
        xyz, scaling, rotation, fdc, opacity, params, rect, counts, N);
    gs_scan_kernel<<<1, 1024, 0, stream>>>(counts, starts, cursors);
    gs_scatter_kernel<<<(N + 255) / 256, 256, 0, stream>>>(rect, cursors,
                                                           list, N);
    gs_render_kernel<<<NBINS / 8, 256, 0, stream>>>(params, counts, starts,
                                                    list, (float*)d_out, N);
}
